// Explainer_51926154608717
// MI455X (gfx1250) — matrix operand order repaired and checked by reading, not compiled, not run on hardware
//
#include <hip/hip_runtime.h>
#include <hip/hip_bf16.h>
#include <math.h>

// Problem constants (from reference)
#define NN   8192      // nodes
#define EE   262144    // edges
#define DD   128       // embed dim (== F == H == K of all GEMMs)
#define CC   7         // classes

typedef __attribute__((ext_vector_type(16))) _Float16 v16h;
typedef __attribute__((ext_vector_type(8)))  float    v8f;

// ---------------------------------------------------------------------------
// WMMA GEMM: C[M,Ncols] = A[M,128] @ B[128,Ncols], f32 in memory, f16
// multiply, f32 accumulate. One wave32 per 16x16 tile; K=128 fully unrolled
// into 4 v_wmma_f32_16x16x32_f16. Layouts per CDNA5 ISA 7.12.2 (wave32):
//   A: lane = M%16, K split by lane-half; halves p<4 -> K+0..7, p>=4 -> K+16..23
//   B: lane = N%16, the 16 f16 halves = K, split by lane-half as for A
//   C/D: lane = N%16, accumulator r -> M = (lane>=16)*8 + r
// nshift = log2(Ncols/16).
// ---------------------------------------------------------------------------
__global__ void gemm_wmma_f16(const float* __restrict__ A,
                              const float* __restrict__ B,
                              float* __restrict__ C,
                              int M, int Ncols, int nshift) {
    const int wave = blockIdx.x * (blockDim.x >> 5) + (threadIdx.x >> 5);
    const int lane = threadIdx.x & 31;
    const int tm = wave >> nshift;
    const int tn = wave & ((1 << nshift) - 1);
    if (tm * 16 >= M) return;                 // wave-uniform: EXEC stays full
    const int m    = tm * 16 + (lane & 15);
    const int half = lane >> 4;

    v8f acc = {};
#pragma unroll
    for (int kk = 0; kk < 128; kk += 32) {
        // ---- A fragment: row m, two aligned runs of 8 consecutive K ----
        const float* arow = A + (size_t)m * DD + kk + half * 8;
        const float4 a0 = *(const float4*)(arow);
        const float4 a1 = *(const float4*)(arow + 4);
        const float4 a2 = *(const float4*)(arow + 16);
        const float4 a3 = *(const float4*)(arow + 20);
        v16h a;
        a[0]=(_Float16)a0.x; a[1]=(_Float16)a0.y; a[2]=(_Float16)a0.z; a[3]=(_Float16)a0.w;
        a[4]=(_Float16)a1.x; a[5]=(_Float16)a1.y; a[6]=(_Float16)a1.z; a[7]=(_Float16)a1.w;
        a[8]=(_Float16)a2.x; a[9]=(_Float16)a2.y; a[10]=(_Float16)a2.z; a[11]=(_Float16)a2.w;
        a[12]=(_Float16)a3.x; a[13]=(_Float16)a3.y; a[14]=(_Float16)a3.z; a[15]=(_Float16)a3.w;
        // ---- B fragment: column n = lane%16, K in same order as A (8h+s, 16+8h+s) ----
        const float* bcol = B + (size_t)(kk + half * 8) * Ncols + (tn << 4) + (lane & 15);
        v16h b;
#pragma unroll
        for (int s = 0; s < 8; ++s) {
            b[s]     = (_Float16)bcol[(size_t)s * Ncols];
            b[8 + s] = (_Float16)bcol[(size_t)(16 + s) * Ncols];
        }
        acc = __builtin_amdgcn_wmma_f32_16x16x32_f16(false, a, false, b,
                                                     (short)0, acc, false, false);
    }
    const int n = (tn << 4) + (lane & 15);
    const int mbase = tm * 16 + half * 8;
#pragma unroll
    for (int r = 0; r < 8; ++r)
        C[(size_t)(mbase + r) * Ncols + n] = acc[r];
}

// cvec[n] = b1e[n] + embed[nodeid] @ W1e[256:384, n]   (constant across edges)
__global__ void cvec_kernel(const float* __restrict__ embed,
                            const float* __restrict__ W1e,
                            const float* __restrict__ b1e,
                            const int* __restrict__ nodeid,
                            float* __restrict__ cvec) {
    const int n = threadIdx.x;                // 64 threads
    const float* er = embed + (size_t)(*nodeid) * DD;
    float s = b1e[n];
    for (int k = 0; k < DD; ++k) s += er[k] * W1e[(size_t)(256 + k) * 64 + n];
    cvec[n] = s;
}

// Zero exactly the mask cells that will be read/written (fwd + reverse).
__global__ void zero_mask_pos(const int* __restrict__ row,
                              const int* __restrict__ col,
                              float* __restrict__ mask) {
    const int e = blockIdx.x * blockDim.x + threadIdx.x;
    if (e >= EE) return;
    const size_t i = (size_t)row[e], j = (size_t)col[e];
    mask[i * NN + j] = 0.0f;
    mask[j * NN + i] = 0.0f;
}

// Per-edge explainer head: la = relu(EA[i]+EB[j]+cvec) @ W2e + b2e,
// concrete-sigmoid gate, scatter into mask[i,j].
__global__ void edge_gate_kernel(const float* __restrict__ EA,
                                 const float* __restrict__ EB,
                                 const float* __restrict__ cvec,
                                 const float* __restrict__ W2e,
                                 const float* __restrict__ b2e,
                                 const float* __restrict__ noise,
                                 const float* __restrict__ tmp,
                                 const int* __restrict__ row,
                                 const int* __restrict__ col,
                                 float* __restrict__ mask) {
    const int e = blockIdx.x * blockDim.x + threadIdx.x;
    if (e >= EE) return;
    const int i = row[e], j = col[e];
    const float4* ea = (const float4*)(EA + (size_t)i * 64);
    const float4* eb = (const float4*)(EB + (size_t)j * 64);
    const float4* cv = (const float4*)cvec;
    const float4* w2 = (const float4*)W2e;
    float la = b2e[0];
#pragma unroll 4
    for (int q = 0; q < 16; ++q) {
        const float4 a = ea[q], b = eb[q], c = cv[q], w = w2[q];
        float h;
        h = a.x + b.x + c.x; la += (h > 0.0f ? h : 0.0f) * w.x;
        h = a.y + b.y + c.y; la += (h > 0.0f ? h : 0.0f) * w.y;
        h = a.z + b.z + c.z; la += (h > 0.0f ? h : 0.0f) * w.z;
        h = a.w + b.w + c.w; la += (h > 0.0f ? h : 0.0f) * w.w;
    }
    const float ns = noise[e];
    const float g = (logf(ns) - log1pf(-ns) + la) / tmp[0];
    const float v = 1.0f / (1.0f + expf(-g));
    atomicAdd(mask + (size_t)i * NN + j, v);
}

__global__ void zero_h1_out(float* __restrict__ h1, float* __restrict__ out_acc) {
    const int t = blockIdx.x * blockDim.x + threadIdx.x;
    if (t < NN * DD) h1[t] = 0.0f;
    if (t < 8) out_acc[t] = 0.0f;
}

// SpMM over edge list: h1[i,:] += adj_data[e] * sym_mask(i,j) * XW[j,:]
// Edge coefficient is block-uniform: compute once, broadcast via LDS.
__global__ void spmm_edge_kernel(const int* __restrict__ row,
                                 const int* __restrict__ col,
                                 const float* __restrict__ adj_data,
                                 const float* __restrict__ mask,
                                 const float* __restrict__ XW,
                                 float* __restrict__ h1) {
    __shared__ float cs;
    __shared__ int   si, sj;
    const int e = blockIdx.x;                 // one block (128 thr) per edge
    if (threadIdx.x == 0) {
        const int i = row[e], j = col[e];
        si = i; sj = j;
        cs = (i == j) ? 0.0f
           : 0.5f * (mask[(size_t)i * NN + j] + mask[(size_t)j * NN + i])
                  * adj_data[e];
    }
    __syncthreads();
    const float c = cs;
    if (c != 0.0f)
        atomicAdd(h1 + (size_t)si * DD + threadIdx.x,
                  c * XW[(size_t)sj * DD + threadIdx.x]);
}

// HW1[i,c] = relu(h1[i,:]) @ Wg1[:,c]
__global__ void hw1_kernel(const float* __restrict__ h1,
                           const float* __restrict__ Wg1,
                           float* __restrict__ HW1) {
    const int idx = blockIdx.x * blockDim.x + threadIdx.x;  // NN*CC
    if (idx >= NN * CC) return;
    const int i = idx / CC, cc = idx - i * CC;
    const float* hr = h1 + (size_t)i * DD;
    float s = 0.0f;
#pragma unroll 8
    for (int d = 0; d < DD; ++d) {
        float h = hr[d];
        s += (h > 0.0f ? h : 0.0f) * Wg1[d * CC + cc];
    }
    HW1[idx] = s;
}

// out_acc[c] = sum over edges from nodeid of c_e * HW1[col, c]
__global__ void final_agg_kernel(const int* __restrict__ row,
                                 const int* __restrict__ col,
                                 const float* __restrict__ adj_data,
                                 const float* __restrict__ mask,
                                 const float* __restrict__ HW1,
                                 const int* __restrict__ nodeid,
                                 float* __restrict__ out_acc) {
    const int e = blockIdx.x * blockDim.x + threadIdx.x;
    if (e >= EE) return;
    const int i = row[e];
    if (i != *nodeid) return;
    const int j = col[e];
    if (i == j) return;
    const float c = 0.5f * (mask[(size_t)i * NN + j] + mask[(size_t)j * NN + i])
                  * adj_data[e];
#pragma unroll
    for (int cc = 0; cc < CC; ++cc)
        atomicAdd(out_acc + cc, c * HW1[(size_t)j * CC + cc]);
}

__global__ void softmax7_kernel(const float* __restrict__ out_acc,
                                float* __restrict__ out) {
    if (threadIdx.x != 0) return;
    float m = out_acc[0];
    for (int c = 1; c < CC; ++c) m = fmaxf(m, out_acc[c]);
    float ex[CC], s = 0.0f;
    for (int c = 0; c < CC; ++c) { ex[c] = expf(out_acc[c] - m); s += ex[c]; }
    for (int c = 0; c < CC; ++c) out[c] = ex[c] / s;
}

extern "C" void kernel_launch(void* const* d_in, const int* in_sizes, int n_in,
                              void* d_out, int out_size, void* d_ws, size_t ws_size,
                              hipStream_t stream) {
    const float* x        = (const float*)d_in[0];
    const float* embed    = (const float*)d_in[1];
    const float* adj_data = (const float*)d_in[2];
    const float* noise    = (const float*)d_in[3];
    const float* tmp      = (const float*)d_in[4];
    const float* W1e      = (const float*)d_in[5];   // [384,64] row-major
    const float* b1e      = (const float*)d_in[6];
    const float* W2e      = (const float*)d_in[7];
    const float* b2e      = (const float*)d_in[8];
    const float* Wg0      = (const float*)d_in[9];   // [128,128]
    const float* Wg1      = (const float*)d_in[10];  // [128,7]
    const int*   row      = (const int*)d_in[11];
    const int*   col      = (const int*)d_in[12];
    const int*   nodeid   = (const int*)d_in[13];
    float* out = (float*)d_out;

    // Workspace layout (needs ~269 MB; mask is only sparsely re-zeroed per call)
    char* p = (char*)d_ws;
    float* mask    = (float*)p; p += (size_t)NN * NN * sizeof(float);   // 256 MB
    float* EA      = (float*)p; p += (size_t)NN * 64 * sizeof(float);   // 2 MB
    float* EB      = (float*)p; p += (size_t)NN * 64 * sizeof(float);   // 2 MB
    float* XW      = (float*)p; p += (size_t)NN * DD * sizeof(float);   // 4 MB
    float* h1      = (float*)p; p += (size_t)NN * DD * sizeof(float);   // 4 MB
    float* HW1     = (float*)p; p += (size_t)NN * CC * sizeof(float);
    float* cvec    = (float*)p; p += 256;
    float* out_acc = (float*)p; p += 256;
    (void)ws_size; (void)n_in; (void)in_sizes; (void)out_size;

    // 1) Node-level WMMA GEMMs (8 waves of 16x16 tiles per 256-thread block)
    //    EA = embed @ W1e[0:128,:]   (8192x128x64)  -> 2048 tiles -> 256 blocks
    //    EB = embed @ W1e[128:256,:] (8192x128x64)
    //    XW = x @ Wg0                (8192x128x128) -> 4096 tiles -> 512 blocks
    gemm_wmma_f16<<<256, 256, 0, stream>>>(embed, W1e,            EA, NN, 64, 2);
    gemm_wmma_f16<<<256, 256, 0, stream>>>(embed, W1e + 128 * 64, EB, NN, 64, 2);
    gemm_wmma_f16<<<512, 256, 0, stream>>>(x,     Wg0,            XW, NN, DD, 3);

    // 2) Constant self-embedding contribution + bias
    cvec_kernel<<<1, 64, 0, stream>>>(embed, W1e, b1e, nodeid, cvec);

    // 3) Sparse-zero touched mask cells, then per-edge gate + scatter
    zero_mask_pos<<<EE / 256, 256, 0, stream>>>(row, col, mask);
    edge_gate_kernel<<<EE / 256, 256, 0, stream>>>(EA, EB, cvec, W2e, b2e,
                                                   noise, tmp, row, col, mask);

    // 4) GCN layer 1 as edge-list SpMM with atomics
    zero_h1_out<<<(NN * DD) / 256, 256, 0, stream>>>(h1, out_acc);
    spmm_edge_kernel<<<EE, DD, 0, stream>>>(row, col, adj_data, mask, XW, h1);

    // 5) relu + second weight, then aggregate only nodeid's row, softmax
    hw1_kernel<<<(NN * CC + 255) / 256, 256, 0, stream>>>(h1, Wg1, HW1);
    final_agg_kernel<<<EE / 256, 256, 0, stream>>>(row, col, adj_data, mask,
                                                   HW1, nodeid, out_acc);
    softmax7_kernel<<<1, 32, 0, stream>>>(out_acc, out);
}
